// CascadeClassifier_89799176225517
// MI455X (gfx1250) — compile-verified, matrix operand order checked by reading, not run on hardware
//
#include <hip/hip_runtime.h>
#include <hip/hip_bf16.h>

typedef __attribute__((ext_vector_type(16))) __bf16 v16bf;
typedef __attribute__((ext_vector_type(8)))  float  v8f;

#define BM 128
#define BN 128
#define BK 32

__device__ __forceinline__ unsigned short f2bf1(float f) {
    unsigned u = __builtin_bit_cast(unsigned, f) + 0x8000u;   // round-half-up
    return (unsigned short)(u >> 16);
}

// pack two fp32 -> bf16x2 (adds + one v_perm_b32)
__device__ __forceinline__ unsigned pk_bf16(float x, float y) {
#if __has_builtin(__builtin_amdgcn_cvt_pk_bf16_f32)
    typedef __attribute__((ext_vector_type(2))) __bf16 v2bf;
    v2bf p = __builtin_amdgcn_cvt_pk_bf16_f32(x, y);
    return __builtin_bit_cast(unsigned, p);
#elif __has_builtin(__builtin_amdgcn_perm)
    unsigned ux = __builtin_bit_cast(unsigned, x) + 0x8000u;
    unsigned uy = __builtin_bit_cast(unsigned, y) + 0x8000u;
    return __builtin_amdgcn_perm(uy, ux, 0x07060302u);
#else
    return (unsigned)f2bf1(x) | ((unsigned)f2bf1(y) << 16);
#endif
}

// C[M,N] = epilogue( A[M,K] @ W[K,N] ), W always fp32.
// ABF: A is bf16 (row-major);  CBF: store C as bf16.
// epilogue: +bias, ReLU, +so2p[gt_s*1001+gt_o][col]*exp(fac) (fp32-C only).
template<bool ABF, bool CBF>
__global__ __launch_bounds__(256)
void fused_gemm_bf16(const void* __restrict__ Av, const float* __restrict__ W,
                     const float* __restrict__ bias, void* __restrict__ Cv,
                     int M, int N, int K, int relu,
                     const int* __restrict__ gt_s, const int* __restrict__ gt_o,
                     const float* __restrict__ so2p, const float* __restrict__ so2p_factor)
{
    __shared__ unsigned short Asl[BM * BK];   // [m][k] bf16
    __shared__ unsigned short Bsl[BN * BK];   // [n][k] bf16 (transposed tile of W)

    const float*          Af = (const float*)Av;
    const unsigned short* Ab = (const unsigned short*)Av;
    float*                Cf = (float*)Cv;
    unsigned short*       Cb = (unsigned short*)Cv;

    const int tid   = threadIdx.x;
    const int lane  = tid & 31;
    const int wave  = tid >> 5;        // 0..7
    const int waveM = wave >> 1;       // 0..3 -> 32 rows each
    const int waveN = wave & 1;        // 0..1 -> 64 cols each
    const int m0 = blockIdx.y * BM;
    const int n0 = blockIdx.x * BN;

    const int laneLo = lane & 15;
    const int kbase  = (lane >> 4) * 8;    // WMMA 16-bit A/B lane K-group

    // A fetch mapping (fp32): 4 float4/thread, rows a_r+32i
    const int a_c4 = tid & 7;
    const int a_r  = tid >> 3;
    // A fetch mapping (bf16): 1 uint4 (8 bf16)/thread
    const int ab_r = tid >> 1;
    const int ab_c = (tid & 1) * 16;
    // B fetch mapping: k-pair rows (b_kp,b_kp+1) and (+16), n-cols b_n..b_n+3
    const int b_n  = (tid & 31) * 4;
    const int b_kp = (tid >> 5) * 2;

    const bool full_tile = ((N & 3) == 0) && (n0 + BN <= N);

    v8f acc[2][4];
#pragma unroll
    for (int mf = 0; mf < 2; ++mf)
#pragma unroll
        for (int nf = 0; nf < 4; ++nf) acc[mf][nf] = (v8f){};

    float4 aR[4];
    uint4  aU;
    float4 bR[2][2];

    auto fetchA = [&](int k0) {
        if constexpr (ABF) {
            aU = *(const uint4*)(Ab + (size_t)(m0 + ab_r) * K + k0 + ab_c);
        } else {
#pragma unroll
            for (int i = 0; i < 4; ++i)
                aR[i] = *(const float4*)(Af + (size_t)(m0 + a_r + i * 32) * K + k0 + a_c4 * 4);
        }
    };
    auto fetchB = [&](int k0) {
        if (full_tile) {
#pragma unroll
            for (int i = 0; i < 2; ++i)
#pragma unroll
                for (int j = 0; j < 2; ++j)
                    bR[i][j] = *(const float4*)(W + (size_t)(k0 + b_kp + i * 16 + j) * N
                                                  + n0 + b_n);
        } else {
#pragma unroll
            for (int i = 0; i < 2; ++i)
#pragma unroll
                for (int j = 0; j < 2; ++j) {
                    const float* wp = W + (size_t)(k0 + b_kp + i * 16 + j) * N;
                    float4 f4;
                    f4.x = (n0 + b_n + 0 < N) ? wp[n0 + b_n + 0] : 0.0f;
                    f4.y = (n0 + b_n + 1 < N) ? wp[n0 + b_n + 1] : 0.0f;
                    f4.z = (n0 + b_n + 2 < N) ? wp[n0 + b_n + 2] : 0.0f;
                    f4.w = (n0 + b_n + 3 < N) ? wp[n0 + b_n + 3] : 0.0f;
                    bR[i][j] = f4;
                }
        }
    };
    auto stageLDS = [&]() {
        if constexpr (ABF) {
            *(uint4*)&Asl[ab_r * BK + ab_c] = aU;
        } else {
#pragma unroll
            for (int i = 0; i < 4; ++i) {
                uint2 p;
                p.x = pk_bf16(aR[i].x, aR[i].y);
                p.y = pk_bf16(aR[i].z, aR[i].w);
                *(uint2*)&Asl[(a_r + i * 32) * BK + a_c4 * 4] = p;
            }
        }
#pragma unroll
        for (int i = 0; i < 2; ++i) {
            const int k = b_kp + i * 16;
#pragma unroll
            for (int j = 0; j < 4; ++j) {
                unsigned v = pk_bf16(((const float*)&bR[i][0])[j],
                                     ((const float*)&bR[i][1])[j]);
                *(unsigned*)&Bsl[(b_n + j) * BK + k] = v;
            }
        }
    };

    fetchA(0); fetchB(0);

    for (int k0 = 0; k0 < K; k0 += BK) {
        stageLDS();
        __syncthreads();

        if (k0 + BK < K) {                // overlap next tile's global loads with WMMA
            if (k0 + 2 * BK < K) {
                if constexpr (ABF)
                    __builtin_prefetch(Ab + (size_t)(m0 + ab_r) * K + (k0 + 2 * BK) + ab_c, 0, 1);
                else
                    __builtin_prefetch(Af + (size_t)(m0 + a_r) * K + (k0 + 2 * BK) + a_c4 * 4, 0, 1);
                __builtin_prefetch(W + (size_t)(k0 + 2 * BK + b_kp) * N + n0 + b_n, 0, 1);
            }
            fetchA(k0 + BK);
            fetchB(k0 + BK);
        }

        // fragments per ISA 16-bit layout: pairs at k = {0,2,4,6,16,18,20,22} + kbase
        union { unsigned u[8]; v16bf v; } afrag[2], bfrag[4];
#pragma unroll
        for (int mf = 0; mf < 2; ++mf) {
            const int row = waveM * 32 + mf * 16 + laneLo;
#pragma unroll
            for (int p = 0; p < 8; ++p) {
                const int k = ((p < 4) ? (2 * p) : (16 + 2 * (p - 4))) + kbase;
                afrag[mf].u[p] = *(const unsigned*)&Asl[row * BK + k];
            }
        }
#pragma unroll
        for (int nf = 0; nf < 4; ++nf) {
            const int col = waveN * 64 + nf * 16 + laneLo;
#pragma unroll
            for (int p = 0; p < 8; ++p) {
                const int k = ((p < 4) ? (2 * p) : (16 + 2 * (p - 4))) + kbase;
                bfrag[nf].u[p] = *(const unsigned*)&Bsl[col * BK + k];
            }
        }
#pragma unroll
        for (int mf = 0; mf < 2; ++mf)
#pragma unroll
            for (int nf = 0; nf < 4; ++nf)
                acc[mf][nf] = __builtin_amdgcn_wmma_f32_16x16x32_bf16(
                    false, afrag[mf].v, false, bfrag[nf].v,
                    (short)0, acc[mf][nf], false, false);

        __syncthreads();
    }

    // ---- epilogue ----
    float fexp = 1.0f;
    if (so2p) fexp = __expf(so2p_factor[0]);
#pragma unroll
    for (int mf = 0; mf < 2; ++mf) {
#pragma unroll
        for (int r = 0; r < 8; ++r) {
            const int row = m0 + waveM * 32 + mf * 16 + r + ((lane >> 4) * 8);
            if (row >= M) continue;
            long so_row = 0;
            if (!CBF && so2p) so_row = ((long)gt_s[row] * 1001 + (long)gt_o[row]) * 132;
#pragma unroll
            for (int nf = 0; nf < 4; ++nf) {
                const int col = n0 + waveN * 64 + nf * 16 + laneLo;
                if (col >= N) continue;
                float v = acc[mf][nf][r];
                if (bias) v += bias[col];
                if (relu) v = fmaxf(v, 0.0f);
                if constexpr (CBF) {
                    Cb[(size_t)row * N + col] = f2bf1(v);
                } else {
                    if (so2p) v += so2p[so_row + col] * fexp;
                    __builtin_nontemporal_store(v, &Cf[(size_t)row * N + col]);
                }
            }
        }
    }
}

__global__ void concat_kernel(const unsigned short* __restrict__ h,
                              const float* __restrict__ ppf,
                              unsigned short* __restrict__ out, int total)
{
    int i = blockIdx.x * blockDim.x + threadIdx.x;
    if (i >= total) return;
    int r = i / 576, c = i % 576;
    out[i] = (c < 512) ? h[(size_t)r * 512 + c] : f2bf1(ppf[(size_t)r * 64 + (c - 512)]);
}

static void launch_gemm(bool abf, bool cbf,
                        const void* A, const float* W, const float* bias, void* C,
                        int M, int N, int K, int relu,
                        const int* gs, const int* go, const float* so, const float* fac,
                        hipStream_t stream)
{
    dim3 blk(256), grd((N + BN - 1) / BN, (M + BM - 1) / BM);
    if (abf && cbf)
        hipLaunchKernelGGL((fused_gemm_bf16<true, true>), grd, blk, 0, stream,
                           A, W, bias, C, M, N, K, relu, gs, go, so, fac);
    else if (abf)
        hipLaunchKernelGGL((fused_gemm_bf16<true, false>), grd, blk, 0, stream,
                           A, W, bias, C, M, N, K, relu, gs, go, so, fac);
    else if (cbf)
        hipLaunchKernelGGL((fused_gemm_bf16<false, true>), grd, blk, 0, stream,
                           A, W, bias, C, M, N, K, relu, gs, go, so, fac);
    else
        hipLaunchKernelGGL((fused_gemm_bf16<false, false>), grd, blk, 0, stream,
                           A, W, bias, C, M, N, K, relu, gs, go, so, fac);
}

extern "C" void kernel_launch(void* const* d_in, const int* in_sizes, int n_in,
                              void* d_out, int out_size, void* d_ws, size_t ws_size,
                              hipStream_t stream)
{
    const float* inp_sf  = (const float*)d_in[0];
    const float* inp_of  = (const float*)d_in[1];
    const float* inp_ppf = (const float*)d_in[2];
    const float* inp_pvf = (const float*)d_in[3];
    const int*   gt_s    = (const int*)d_in[4];
    const int*   gt_o    = (const int*)d_in[5];
    const float* W_obj1  = (const float*)d_in[6];
    const float* b_obj1  = (const float*)d_in[7];
    const float* W_obj2  = (const float*)d_in[8];
    const float* W_pvf   = (const float*)d_in[9];
    const float* b_pvf   = (const float*)d_in[10];
    const float* W_pf    = (const float*)d_in[11];
    const float* b_pf    = (const float*)d_in[12];
    const float* W_pred  = (const float*)d_in[13];
    const float* so2p    = (const float*)d_in[14];
    const float* so2p_f  = (const float*)d_in[15];
    float* out = (float*)d_out;

    const int B = in_sizes[0] / 1024;
    const int OBJ_FEAT = 1024, HIDDEN = 512, OBJ_NUM = 1001, CATK = 576, PRED_NUM = 132;

    unsigned short* ws_h   = (unsigned short*)d_ws;          // B*512 bf16
    unsigned short* ws_cat = ws_h + (size_t)B * HIDDEN;      // B*576 bf16

    // subject scores
    launch_gemm(false, true, inp_sf, W_obj1, b_obj1, ws_h, B, HIDDEN, OBJ_FEAT, 1,
                nullptr, nullptr, nullptr, nullptr, stream);
    launch_gemm(true, false, ws_h, W_obj2, nullptr, out, B, OBJ_NUM, HIDDEN, 0,
                nullptr, nullptr, nullptr, nullptr, stream);
    // object scores
    launch_gemm(false, true, inp_of, W_obj1, b_obj1, ws_h, B, HIDDEN, OBJ_FEAT, 1,
                nullptr, nullptr, nullptr, nullptr, stream);
    launch_gemm(true, false, ws_h, W_obj2, nullptr, out + (size_t)B * OBJ_NUM,
                B, OBJ_NUM, HIDDEN, 0, nullptr, nullptr, nullptr, nullptr, stream);
    // predicate path
    launch_gemm(false, true, inp_pvf, W_pvf, b_pvf, ws_h, B, HIDDEN, OBJ_FEAT, 1,
                nullptr, nullptr, nullptr, nullptr, stream);
    {
        int total = B * CATK;
        hipLaunchKernelGGL(concat_kernel, dim3((total + 255) / 256), dim3(256), 0, stream,
                           ws_h, inp_ppf, ws_cat, total);
    }
    launch_gemm(true, true, ws_cat, W_pf, b_pf, ws_h, B, HIDDEN, CATK, 1,
                nullptr, nullptr, nullptr, nullptr, stream);
    launch_gemm(true, false, ws_h, W_pred, nullptr, out + (size_t)2 * B * OBJ_NUM,
                B, PRED_NUM, HIDDEN, 0, gt_s, gt_o, so2p, so2p_f, stream);
}
